// OptimizedAttention_57655640982115
// MI455X (gfx1250) — matrix operand order repaired and checked by reading, not compiled, not run on hardware
//
#include <hip/hip_runtime.h>
#include <hip/hip_bf16.h>

// MI455X (gfx1250) fused RoPE-attention, bf16 WMMA with fp32 accumulate.
// Compute-bound (~770 FLOP/byte): V_WMMA_F32_16X16X32_BF16 everywhere,
// LDS-staged B tiles (convert-once), async global->LDS staging of K/V.

typedef __attribute__((ext_vector_type(16))) __bf16 v16bf;
typedef __attribute__((ext_vector_type(8)))  __bf16 v8bf;
typedef __attribute__((ext_vector_type(8)))  float  v8f;
typedef int v4i __attribute__((vector_size(16)));   // matches async-LDS builtin

constexpr int EMBED = 2048;
constexpr int SEQ   = 512;
constexpr int BATCH = 4;
constexpr int HEADS = 16;
constexpr int HDIM  = 128;
constexpr int NQKV  = 3 * EMBED;
constexpr float kScale = 0.08838834764831845f;  // 1/sqrt(128)
#define NEG_BIG (-3.0e38f)

#define AS1 __attribute__((address_space(1)))
#define AS3 __attribute__((address_space(3)))

#if __has_builtin(__builtin_amdgcn_global_load_async_to_lds_b128)
#define HAVE_ASYNC_LDS 1
#else
#define HAVE_ASYNC_LDS 0
#endif

// 16-byte global -> LDS copy; async (ASYNCcnt) when the builtin exists.
__device__ __forceinline__ void async_cp16(const __bf16* g, __bf16* l) {
#if HAVE_ASYNC_LDS
  __builtin_amdgcn_global_load_async_to_lds_b128(
      (AS1 v4i*)(uintptr_t)g, (AS3 v4i*)(uintptr_t)l, 0, 0);
#else
  *(v8bf*)l = *(const v8bf*)g;
#endif
}

__device__ __forceinline__ void wait_async() {
#if HAVE_ASYNC_LDS
  asm volatile("s_wait_asynccnt 0" ::: "memory");
#endif
}

__device__ __forceinline__ v8bf cvt8(const float* p) {
  const float4 f0 = ((const float4*)p)[0];
  const float4 f1 = ((const float4*)p)[1];
  v8bf r;
  r[0] = (__bf16)f0.x; r[1] = (__bf16)f0.y; r[2] = (__bf16)f0.z; r[3] = (__bf16)f0.w;
  r[4] = (__bf16)f1.x; r[5] = (__bf16)f1.y; r[6] = (__bf16)f1.z; r[7] = (__bf16)f1.w;
  return r;
}

__device__ __forceinline__ v16bf cat8(v8bf lo, v8bf hi) {
  return __builtin_shufflevector(lo, hi, 0,1,2,3,4,5,6,7,8,9,10,11,12,13,14,15);
}

__device__ __forceinline__ v8f wmma_bf16(v16bf a, v16bf b, v8f c) {
  // D = A(16x32 bf16) * B(32x16 bf16) + C(16x16 f32)
  return __builtin_amdgcn_wmma_f32_16x16x32_bf16(false, a, false, b, (short)0, c,
                                                 false, false);
}

// ---------------------------------------------------------------------------
// Kernel 1: qkv = x @ w_qkv^T with RoPE epilogue on q,k.
// B tile (64N x 32K) converted to bf16 once per block into double-buffered
// LDS; each of the 8 waves computes a 16(M) x 64(N) tile.
// q,k -> bf16 [B,H,S,Hd];  v -> bf16 transposed [B,H,Hd,S].
// ---------------------------------------------------------------------------
__global__ __launch_bounds__(256) void qkv_rope_wmma(
    const float* __restrict__ x, const float* __restrict__ wqkv,
    __bf16* __restrict__ qws, __bf16* __restrict__ kws,
    __bf16* __restrict__ vtws) {
  __shared__ alignas(32) __bf16 ldsB[2][64 * 32];
  const int tid  = threadIdx.x;
  const int lane = tid & 31;
  const int wave = tid >> 5;
  const int half = lane >> 4;   // K-offset selector per WMMA layout tables
  const int l16  = lane & 15;
  const int mBase = blockIdx.y * 128 + wave * 16;
  const int nBase = blockIdx.x * 64;

  // cooperative B staging: thread -> (row sn, 8-wide K chunk sk)
  const int sn = tid >> 2;
  const int sk = (tid & 3) * 8;
  const float* wsrc = wqkv + (size_t)(nBase + sn) * EMBED + sk;

  v8f acc[4] = {};
  const float* xrow = x + (size_t)(mBase + l16) * EMBED + half * 8;

  int p = 0;
  for (int kb = 0; kb < EMBED; kb += 32, p ^= 1) {
    *(v8bf*)&ldsB[p][sn * 32 + sk] = cvt8(wsrc + kb);   // convert once/block
    __builtin_prefetch(wsrc + kb + 256, 0, 1);
    __syncthreads();
    // A 16x32: elems 0..7 = K kb+8h.. ; elems 8..15 = K kb+8h+16..
    const v16bf a = cat8(cvt8(xrow + kb), cvt8(xrow + kb + 16));
    v16bf bfr[4];                     // preload all B frags -> batch ds_loads
#pragma unroll
    for (int t = 0; t < 4; ++t)
      bfr[t] = cat8(*(const v8bf*)&ldsB[p][(t * 16 + l16) * 32 + half * 8],
                    *(const v8bf*)&ldsB[p][(t * 16 + l16) * 32 + half * 8 + 16]);
#pragma unroll
    for (int t = 0; t < 4; ++t) acc[t] = wmma_bf16(a, bfr[t], acc[t]);
  }

  const int m0 = mBase + half * 8;     // element r of acc holds row m0+r
  const int bb = m0 / SEQ;
  const int s0 = m0 % SEQ;

#pragma unroll
  for (int t = 0; t < 4; ++t) {
    const int n    = nBase + t * 16 + l16;
    const int part = n / EMBED;        // 0=q 1=k 2=v (uniform over the tile)
    const int d    = n % EMBED;
    const int h    = d / HDIM;         // uniform over the tile
    const int hd   = d % HDIM;
    const size_t bh = (size_t)bb * HEADS + h;
    if (part == 2) {
      v8bf vv;                         // V^T: 8 row values contiguous in S
#pragma unroll
      for (int r = 0; r < 8; ++r) vv[r] = (__bf16)acc[t][r];
      *(v8bf*)(vtws + (bh * HDIM + hd) * SEQ + s0) = vv;
    } else {
      __bf16* dst = (part == 0 ? qws : kws) + (bh * SEQ + s0) * HDIM + hd;
      const float invf = __powf(10000.0f, -(float)(hd & ~1) * (1.0f / HDIM));
      const bool even = (hd & 1) == 0;
#pragma unroll
      for (int r = 0; r < 8; ++r) {
        const float val   = acc[t][r];
        const float other = __shfl_xor(val, 1, 32);  // hd partner is lane^1
        float sv, cv;
        __sincosf((float)(s0 + r) * invf, &sv, &cv);
        const float res = even ? (val * cv - other * sv)
                               : (other * sv + val * cv);
        dst[(size_t)r * HDIM] = (__bf16)res;
      }
    }
  }
}

// ---------------------------------------------------------------------------
// Kernel 2: causal flash attention per (b,h). Block owns 128 q rows (one
// 16-row tile per wave). All waves run a uniform key loop to the block's max
// causal horizon, cooperatively async-staging the 32x128 K tile and 128x32
// V^T tile into double-buffered LDS; waves past their own horizon only stage.
// Online softmax; P transposed through per-wave LDS.
// ---------------------------------------------------------------------------
__global__ __launch_bounds__(256) void flash_attn_wmma(
    const __bf16* __restrict__ qws, const __bf16* __restrict__ kws,
    const __bf16* __restrict__ vtws, __bf16* __restrict__ aout) {
  __shared__ alignas(32) __bf16 ldsK[2][32 * 128];  // [key][hd]
  __shared__ alignas(32) __bf16 ldsV[2][128 * 32];  // [hd][key]
  __shared__ alignas(32) __bf16 ldsP[8][16 * 32];   // per-wave [q][key]
  const int tid  = threadIdx.x;
  const int lane = tid & 31;
  const int wave = tid >> 5;
  const int half = lane >> 4;
  const int l16  = lane & 15;
  const int bh = blockIdx.x;
  const int bb = bh >> 4;
  const int h  = bh & 15;
  const int qBase = blockIdx.y * 128;
  const int qTile = qBase + wave * 16;

  const __bf16* qb  = qws  + (size_t)bh * SEQ * HDIM;
  const __bf16* kbp = kws  + (size_t)bh * SEQ * HDIM;
  const __bf16* vb  = vtws + (size_t)bh * HDIM * SEQ;

  // Q fragments: 4 K-steps of 32 covering Hd=128, kept resident
  v16bf qf[4];
  {
    const __bf16* qrow = qb + (size_t)(qTile + l16) * HDIM + half * 8;
#pragma unroll
    for (int st = 0; st < 4; ++st)
      qf[st] = cat8(*(const v8bf*)(qrow + st * 32),
                    *(const v8bf*)(qrow + st * 32 + 16));
  }

  float mrow[8], lrow[8], alphas[8];
  v8f o[8] = {};
#pragma unroll
  for (int r = 0; r < 8; ++r) { mrow[r] = NEG_BIG; lrow[r] = 0.0f; }

  const int kEnd      = qTile + 16;   // this wave's causal horizon
  const int blockKEnd = qBase + 128;  // block's max horizon (uniform loop)
  const int c0 = tid, c1 = tid + 256;

  int p = 0;
  for (int j0 = 0; j0 < blockKEnd; j0 += 32, p ^= 1) {
    // cooperative async staging: K 32x128 (8KB) + V^T 128x32 (8KB)
    async_cp16(kbp + (size_t)(j0 + (c0 >> 4)) * HDIM + (c0 & 15) * 8,
               &ldsK[p][(c0 >> 4) * 128 + (c0 & 15) * 8]);
    async_cp16(kbp + (size_t)(j0 + (c1 >> 4)) * HDIM + (c1 & 15) * 8,
               &ldsK[p][(c1 >> 4) * 128 + (c1 & 15) * 8]);
    async_cp16(vb + (size_t)(c0 >> 2) * SEQ + j0 + (c0 & 3) * 8,
               &ldsV[p][(c0 >> 2) * 32 + (c0 & 3) * 8]);
    async_cp16(vb + (size_t)(c1 >> 2) * SEQ + j0 + (c1 & 3) * 8,
               &ldsV[p][(c1 >> 2) * 32 + (c1 & 3) * 8]);
    wait_async();
    __syncthreads();

    if (j0 < kEnd) {  // wave-uniform branch: EXEC stays all-ones inside
      v8f sA = {}, sB = {};
      {
        v16bf kf[4];
#pragma unroll
        for (int st = 0; st < 4; ++st)
          kf[st] =
              cat8(*(const v8bf*)&ldsK[p][l16 * 128 + half * 8 + st * 32],
                   *(const v8bf*)&ldsK[p][l16 * 128 + half * 8 + st * 32 + 16]);
#pragma unroll
        for (int st = 0; st < 4; ++st) sA = wmma_bf16(qf[st], kf[st], sA);
      }
      {
        v16bf kf[4];
#pragma unroll
        for (int st = 0; st < 4; ++st)
          kf[st] =
              cat8(*(const v8bf*)&ldsK[p][(16 + l16) * 128 + half * 8 + st * 32],
                   *(const v8bf*)&ldsK[p][(16 + l16) * 128 + half * 8 + st * 32 + 16]);
#pragma unroll
        for (int st = 0; st < 4; ++st) sB = wmma_bf16(qf[st], kf[st], sB);
      }

      // online softmax: each q row lives in one 16-lane half-wave
#pragma unroll
      for (int r = 0; r < 8; ++r) {
        const int q = qTile + half * 8 + r;
        float a = (j0 + l16      <= q) ? sA[r] * kScale : NEG_BIG;
        float c = (j0 + 16 + l16 <= q) ? sB[r] * kScale : NEG_BIG;
        float mx = fmaxf(a, c);
#pragma unroll
        for (int off = 1; off < 16; off <<= 1)
          mx = fmaxf(mx, __shfl_xor(mx, off, 32));
        const float mnew = fmaxf(mrow[r], mx);
        alphas[r] = __expf(mrow[r] - mnew);
        const float ea = __expf(a - mnew);   // underflows to 0 when masked
        const float eb = __expf(c - mnew);
        float sum = ea + eb;
#pragma unroll
        for (int off = 1; off < 16; off <<= 1) sum += __shfl_xor(sum, off, 32);
        mrow[r] = mnew;
        lrow[r] = lrow[r] * alphas[r] + sum;
        ldsP[wave][(half * 8 + r) * 32 + l16]      = (__bf16)ea;
        ldsP[wave][(half * 8 + r) * 32 + 16 + l16] = (__bf16)eb;
      }
      // same-wave LDS RAW: DS ops are in-order per wave; fence + wait
      asm volatile("s_wait_dscnt 0" ::: "memory");

#pragma unroll
      for (int t = 0; t < 8; ++t)
#pragma unroll
        for (int r = 0; r < 8; ++r) o[t][r] *= alphas[r];

      const v16bf pf =
          cat8(*(const v8bf*)&ldsP[wave][l16 * 32 + half * 8],
               *(const v8bf*)&ldsP[wave][l16 * 32 + half * 8 + 16]);
#pragma unroll
      for (int t = 0; t < 8; ++t)
        o[t] = wmma_bf16(
            pf, cat8(*(const v8bf*)&ldsV[p][(t * 16 + l16) * 32 + half * 8],
                     *(const v8bf*)&ldsV[p][(t * 16 + l16) * 32 + half * 8 + 16]),
            o[t]);
    }
  }

#pragma unroll
  for (int r = 0; r < 8; ++r) lrow[r] = 1.0f / lrow[r];
  __bf16* orow =
      aout + ((size_t)(bb * SEQ + qTile + half * 8)) * EMBED + h * HDIM;
#pragma unroll
  for (int t = 0; t < 8; ++t) {
    const int hd = t * 16 + l16;
#pragma unroll
    for (int r = 0; r < 8; ++r)
      orow[(size_t)r * EMBED + hd] = (__bf16)(o[t][r] * lrow[r]);
  }
}

// ---------------------------------------------------------------------------
// Kernel 3: out = attn @ w_o^T (A already bf16; B staged/converted via LDS)
// ---------------------------------------------------------------------------
__global__ __launch_bounds__(256) void out_proj_wmma(
    const __bf16* __restrict__ ain, const float* __restrict__ wo,
    float* __restrict__ out) {
  __shared__ alignas(32) __bf16 ldsB[2][64 * 32];
  const int tid  = threadIdx.x;
  const int lane = tid & 31;
  const int wave = tid >> 5;
  const int half = lane >> 4;
  const int l16  = lane & 15;
  const int mBase = blockIdx.y * 128 + wave * 16;
  const int nBase = blockIdx.x * 64;

  const int sn = tid >> 2;
  const int sk = (tid & 3) * 8;
  const float* wsrc = wo + (size_t)(nBase + sn) * EMBED + sk;

  v8f acc[4] = {};
  const __bf16* arow = ain + (size_t)(mBase + l16) * EMBED + half * 8;

  int p = 0;
  for (int kb = 0; kb < EMBED; kb += 32, p ^= 1) {
    *(v8bf*)&ldsB[p][sn * 32 + sk] = cvt8(wsrc + kb);
    __builtin_prefetch(wsrc + kb + 256, 0, 1);
    __syncthreads();
    const v16bf a = cat8(*(const v8bf*)(arow + kb),
                         *(const v8bf*)(arow + kb + 16));
    v16bf bfr[4];
#pragma unroll
    for (int t = 0; t < 4; ++t)
      bfr[t] = cat8(*(const v8bf*)&ldsB[p][(t * 16 + l16) * 32 + half * 8],
                    *(const v8bf*)&ldsB[p][(t * 16 + l16) * 32 + half * 8 + 16]);
#pragma unroll
    for (int t = 0; t < 4; ++t) acc[t] = wmma_bf16(a, bfr[t], acc[t]);
  }
  float* orow = out + (size_t)(mBase + half * 8) * EMBED + nBase + l16;
#pragma unroll
  for (int t = 0; t < 4; ++t)
#pragma unroll
    for (int r = 0; r < 8; ++r)
      orow[(size_t)r * EMBED + t * 16] = acc[t][r];
}

// ---------------------------------------------------------------------------
extern "C" void kernel_launch(void* const* d_in, const int* in_sizes, int n_in,
                              void* d_out, int out_size, void* d_ws,
                              size_t ws_size, hipStream_t stream) {
  const float* x    = (const float*)d_in[0];
  const float* wqkv = (const float*)d_in[1];
  const float* wo   = (const float*)d_in[2];
  float* out = (float*)d_out;

  // Workspace carve (bf16): q, k, vT, attn-out = 4 x 8 MB = 32 MB total
  const size_t ELEMS = (size_t)BATCH * SEQ * EMBED;  // 4,194,304
  __bf16* qws  = (__bf16*)d_ws;
  __bf16* kws  = qws + ELEMS;
  __bf16* vtws = kws + ELEMS;
  __bf16* aws  = vtws + ELEMS;

  qkv_rope_wmma<<<dim3(NQKV / 64, (BATCH * SEQ) / 128), 256, 0, stream>>>(
      x, wqkv, qws, kws, vtws);
  flash_attn_wmma<<<dim3(BATCH * HEADS, SEQ / 128), 256, 0, stream>>>(
      qws, kws, vtws, aws);
  out_proj_wmma<<<dim3(EMBED / 64, (BATCH * SEQ) / 128), 256, 0, stream>>>(
      aws, wo, out);
}
